// TransformerEncoderLayer_33947421507649
// MI455X (gfx1250) — matrix operand order repaired and checked by reading, not compiled, not run on hardware
//
#include <hip/hip_runtime.h>
#include <hip/hip_bf16.h>

// ---------------------------------------------------------------------------
// Problem constants (from reference): B=16, L=1024, DM=512, H=8, DH=64,
// DFF=2048, U=35, M_TOT = B*L = 16384.
// ---------------------------------------------------------------------------
typedef _Float16 h16;
typedef __attribute__((ext_vector_type(16))) _Float16 v16h;
typedef __attribute__((ext_vector_type(8)))  float    v8f;

#define C_B    16
#define C_L    1024
#define C_DM   512
#define C_H    8
#define C_DH   64
#define C_DFF  2048
#define C_U    35
#define C_MTOT 16384   // B*L

// ---------------------------------------------------------------------------
// Workspace layout (bytes). Peak ~212 MiB with reuse.
// ---------------------------------------------------------------------------
static constexpr size_t OFF_XH      = 0;                                   // f16 x
static constexpr size_t OFF_WH      = OFF_XH      + (size_t)C_MTOT*C_DM*2; // f16 weight slot (2 MiB)
static constexpr size_t OFF_Q       = OFF_WH      + (size_t)C_DFF*C_DM*2;
static constexpr size_t OFF_K       = OFF_Q       + (size_t)C_MTOT*C_DM*4;
static constexpr size_t OFF_V       = OFF_K       + (size_t)C_MTOT*C_DM*4;
static constexpr size_t OFF_M       = OFF_V       + (size_t)C_MTOT*C_DM*4;
static constexpr size_t OFF_MTOP    = OFF_M       + (size_t)C_B*C_H*C_L*4;
static constexpr size_t OFF_UPD     = OFF_MTOP    + ((size_t)C_B*C_H*C_U*4 + 255ull & ~255ull);
static constexpr size_t OFF_VMEAN   = OFF_UPD     + ((size_t)C_B*C_H*C_U*C_DH*4 + 255ull & ~255ull);
static constexpr size_t OFF_CTXH    = OFF_VMEAN   + ((size_t)C_B*C_H*C_DH*4 + 255ull & ~255ull);
static constexpr size_t OFF_ATTNOUT = OFF_CTXH    + (size_t)C_MTOT*C_DM*2;
static constexpr size_t OFF_X1      = OFF_ATTNOUT + (size_t)C_MTOT*C_DM*4;
static constexpr size_t OFF_X1H     = OFF_X1      + (size_t)C_MTOT*C_DM*4;
static constexpr size_t WS_TOTAL    = OFF_X1H     + (size_t)C_MTOT*C_DM*2;
// Reused regions:
//   yh (f16, M x DFF)      -> OFF_Q      ffn2out (f32) -> OFF_V
//   x2 (f32)               -> OFF_ATTNOUT
//   Aconv (f16, M x 1536)  -> OFF_Q      convout (f32) -> OFF_X1

// ---------------------------------------------------------------------------
// WMMA GEMM: C(M x N) = A(M x K) @ B(K x N) (+bias) [+relu]
// A f16 (M,K) row-major; Bt f16 is B TRANSPOSED: (N,K) row-major.
// Cf f32 (nullable); Ch f16 copy (nullable).
// Block tile 128x128, 8 waves, wave tile 32x64 (2x4 WMMA frags).
// Double-buffered LDS, one barrier per K-step. K % 32 == 0.
// ---------------------------------------------------------------------------
#define BLK_M 128
#define BLK_N 128
#define BLK_K 32

union frag16 { v16h v; struct { uint4 lo, hi; } u; };

__global__ __launch_bounds__(256) void k_gemm(
    const h16* __restrict__ A, const h16* __restrict__ Bt,
    const float* __restrict__ bias, float* __restrict__ Cf,
    h16* __restrict__ Ch, int N, int K, int relu)
{
  __shared__ h16 As[2][BLK_M][BLK_K];   // 2 x 8 KiB
  __shared__ h16 Bs[2][BLK_N][BLK_K];   // 2 x 8 KiB, [n][k]

  const int tid  = threadIdx.x;
  const int lane = tid & 31;
  const int wv   = tid >> 5;
  const int bm   = blockIdx.x * BLK_M;
  const int bn   = blockIdx.y * BLK_N;
  const int waveM = (wv & 3) * 32;   // 4 waves along M
  const int waveN = (wv >> 2) * 64;  // 2 waves along N
  const int row = lane & 15;
  const int g   = lane >> 4;

  // staging address components (each thread moves 16 halves of A and of B)
  const int sr = tid >> 1;           // 0..127 row (A) / col (Bt)
  const int sc = (tid & 1) * 16;     // 0 or 16 (halves within 32-wide K slab)
  const h16* pA = A  + (size_t)(bm + sr) * K + sc;
  const h16* pB = Bt + (size_t)(bn + sr) * K + sc;

  v8f acc[2][4];
#pragma unroll
  for (int mi = 0; mi < 2; ++mi)
#pragma unroll
    for (int ni = 0; ni < 4; ++ni) acc[mi][ni] = (v8f){};

  const int nsteps = K / BLK_K;

  // prologue: stage tile 0
  {
    const uint4* a = (const uint4*)(pA);
    const uint4* b = (const uint4*)(pB);
    uint4 a0 = a[0], a1 = a[1], b0 = b[0], b1 = b[1];
    *(uint4*)(&As[0][sr][sc])     = a0;
    *(uint4*)(&As[0][sr][sc + 8]) = a1;
    *(uint4*)(&Bs[0][sr][sc])     = b0;
    *(uint4*)(&Bs[0][sr][sc + 8]) = b1;
  }
  __syncthreads();

  for (int s = 0; s < nsteps; ++s) {
    const int cur = s & 1, nxt = cur ^ 1;
    uint4 na0, na1, nb0, nb1;
    const bool more = (s + 1 < nsteps);
    if (more) {
      const uint4* a = (const uint4*)(pA + (size_t)(s + 1) * BLK_K);
      const uint4* b = (const uint4*)(pB + (size_t)(s + 1) * BLK_K);
      na0 = a[0]; na1 = a[1]; nb0 = b[0]; nb1 = b[1];
    }
    if (s + 2 < nsteps) {
      __builtin_prefetch(pA + (size_t)(s + 2) * BLK_K, 0, 3);
      __builtin_prefetch(pB + (size_t)(s + 2) * BLK_K, 0, 3);
    }

    // ---- compute 8 WMMAs from buffer `cur` ----
    frag16 fa[2], fb[4];
#pragma unroll
    for (int mi = 0; mi < 2; ++mi) {
      const h16* p = &As[cur][waveM + mi * 16 + row][0];
      fa[mi].u.lo = *(const uint4*)(p + g * 8);
      fa[mi].u.hi = *(const uint4*)(p + 16 + g * 8);
    }
#pragma unroll
    for (int ni = 0; ni < 4; ++ni) {
      const h16* p = &Bs[cur][waveN + ni * 16 + row][0];
      fb[ni].u.lo = *(const uint4*)(p + g * 8);
      fb[ni].u.hi = *(const uint4*)(p + 16 + g * 8);
    }
#pragma unroll
    for (int mi = 0; mi < 2; ++mi)
#pragma unroll
      for (int ni = 0; ni < 4; ++ni)
        acc[mi][ni] = __builtin_amdgcn_wmma_f32_16x16x32_f16(
            false, fa[mi].v, false, fb[ni].v, (short)0, acc[mi][ni], false, false);

    // ---- write next tile into the other buffer ----
    if (more) {
      *(uint4*)(&As[nxt][sr][sc])     = na0;
      *(uint4*)(&As[nxt][sr][sc + 8]) = na1;
      *(uint4*)(&Bs[nxt][sr][sc])     = nb0;
      *(uint4*)(&Bs[nxt][sr][sc + 8]) = nb1;
    }
    __syncthreads();
  }

  // --- epilogue: C layout lane: N = lane&15, VGPR r -> M = r + 8*(lane>>4) ---
  const int col = lane & 15;
#pragma unroll
  for (int mi = 0; mi < 2; ++mi)
#pragma unroll
    for (int ni = 0; ni < 4; ++ni)
#pragma unroll
      for (int r = 0; r < 8; ++r) {
        size_t gr = (size_t)(bm + waveM + mi * 16 + r + 8 * g);
        int    gc = bn + waveN + ni * 16 + col;
        float v = acc[mi][ni][r];
        if (bias) v += bias[gc];
        if (relu) v = fmaxf(v, 0.0f);
        if (Cf) Cf[gr * N + gc] = v;
        if (Ch) Ch[gr * N + gc] = (h16)v;
      }
}

// ---------------------------------------------------------------------------
// Elementwise / attention helper kernels
// ---------------------------------------------------------------------------
__global__ void k_cvt(const float* __restrict__ s, h16* __restrict__ d, size_t n) {
  size_t i = (size_t)blockIdx.x * 256 + threadIdx.x;
  if (i < n) d[i] = (h16)s[i];
}

// dst(C,R) f16 = transpose of src(R,C) f32   (used to build Bt = W^T)
__global__ void k_cvt_tr(const float* __restrict__ s, h16* __restrict__ d, int R, int C) {
  size_t i = (size_t)blockIdx.x * 256 + threadIdx.x;
  if (i >= (size_t)R * C) return;
  int r = (int)(i / C), c = (int)(i % C);
  d[(size_t)c * R + r] = (h16)s[i];
}

// down_w (o,i,t) -> Bt[n=o][k=t*DM+i]
__global__ void k_cvt_downw(const float* __restrict__ s, h16* __restrict__ d) {
  size_t i = (size_t)blockIdx.x * 256 + threadIdx.x;
  if (i >= (size_t)C_DM * C_DM * 3) return;
  int o = (int)(i / (C_DM * 3));
  int rem = (int)(i % (C_DM * 3));
  int ii = rem / 3, t = rem % 3;
  d[(size_t)o * (3 * C_DM) + t * C_DM + ii] = (h16)s[i];
}

// Probe scores: M[b,h,l] = max_u(q.k_samp) - sum_u(q.k_samp)/L
__global__ __launch_bounds__(256) void k_probe(
    const float* __restrict__ Q, const float* __restrict__ Kb,
    const int* __restrict__ idx, float* __restrict__ Mout)
{
  int t = blockIdx.x * 256 + threadIdx.x;
  if (t >= C_B * C_H * C_L) return;
  int l = t & (C_L - 1);
  int bh = t >> 10;
  int b = bh >> 3, h = bh & 7;
  const float* qp = Q + ((size_t)(b * C_L + l)) * C_DM + h * C_DH;
  float q[C_DH];
  for (int d = 0; d < C_DH; ++d) q[d] = qp[d];
  float mx = -3.4e38f, sm = 0.0f;
  for (int u = 0; u < C_U; ++u) {
    int kl = idx[l * C_U + u];
    const float* kp = Kb + ((size_t)(b * C_L + kl)) * C_DM + h * C_DH;
    float s = 0.0f;
    for (int d = 0; d < C_DH; ++d) s += q[d] * kp[d];
    mx = fmaxf(mx, s);
    sm += s;
  }
  Mout[(size_t)bh * C_L + l] = mx - sm * (1.0f / C_L);
}

// Iterative top-U argmax per (b,h)
__global__ __launch_bounds__(256) void k_topk(const float* __restrict__ Mb, int* __restrict__ mtop) {
  __shared__ float vals[C_L];
  __shared__ float rv[256];
  __shared__ int   ri[256];
  int bh = blockIdx.x, tid = threadIdx.x;
  const float* src = Mb + (size_t)bh * C_L;
  for (int i = tid; i < C_L; i += 256) vals[i] = src[i];
  __syncthreads();
  for (int u = 0; u < C_U; ++u) {
    float bv = -3.4e38f; int bi = 0;
    for (int i = tid; i < C_L; i += 256) {
      float v = vals[i];
      if (v > bv) { bv = v; bi = i; }
    }
    rv[tid] = bv; ri[tid] = bi;
    __syncthreads();
    for (int s = 128; s > 0; s >>= 1) {
      if (tid < s && rv[tid + s] > rv[tid]) { rv[tid] = rv[tid + s]; ri[tid] = ri[tid + s]; }
      __syncthreads();
    }
    if (tid == 0) { mtop[bh * C_U + u] = ri[0]; vals[ri[0]] = -3.4e38f; }
    __syncthreads();
  }
}

// Per (b,h,u): scores over L, softmax, upd = attn @ V
__global__ __launch_bounds__(256) void k_attn(
    const float* __restrict__ Q, const float* __restrict__ Kb,
    const float* __restrict__ Vb, const int* __restrict__ mtop,
    float* __restrict__ upd)
{
  __shared__ float qs[C_DH];
  __shared__ float sc[C_L];
  __shared__ float red[256];
  __shared__ float pacc[256];
  int bid = blockIdx.x;
  int u = bid % C_U, bh = bid / C_U;
  int b = bh >> 3, h = bh & 7;
  int tid = threadIdx.x;
  int lsel = mtop[bh * C_U + u];
  size_t baseBL = (size_t)b * C_L;
  if (tid < C_DH) qs[tid] = Q[(baseBL + lsel) * C_DM + h * C_DH + tid];
  __syncthreads();
  const float scale = 0.125f;  // 1/sqrt(64)
  float lmax = -3.4e38f;
  for (int l = tid; l < C_L; l += 256) {
    const float* kp = Kb + (baseBL + l) * C_DM + h * C_DH;
    float s = 0.0f;
    for (int d = 0; d < C_DH; ++d) s += qs[d] * kp[d];
    s *= scale;
    sc[l] = s;
    lmax = fmaxf(lmax, s);
  }
  red[tid] = lmax; __syncthreads();
  for (int s = 128; s > 0; s >>= 1) { if (tid < s) red[tid] = fmaxf(red[tid], red[tid + s]); __syncthreads(); }
  float mx = red[0]; __syncthreads();
  float lsum = 0.0f;
  for (int l = tid; l < C_L; l += 256) { float e = expf(sc[l] - mx); sc[l] = e; lsum += e; }
  red[tid] = lsum; __syncthreads();
  for (int s = 128; s > 0; s >>= 1) { if (tid < s) red[tid] += red[tid + s]; __syncthreads(); }
  float tot = red[0]; __syncthreads();
  int d = tid & 63, ch = tid >> 6;
  float acc = 0.0f;
  for (int l = ch * 256; l < ch * 256 + 256; ++l)
    acc += sc[l] * Vb[(baseBL + l) * C_DM + h * C_DH + d];
  pacc[tid] = acc; __syncthreads();
  if (tid < C_DH)
    upd[((size_t)(bh * C_U + u)) * C_DH + tid] =
        (pacc[tid] + pacc[64 + tid] + pacc[128 + tid] + pacc[192 + tid]) / tot;
}

__global__ void k_vmean(const float* __restrict__ Vb, float* __restrict__ vm) {
  int bh = blockIdx.x, d = threadIdx.x;  // 64 threads
  int b = bh >> 3, h = bh & 7;
  float s = 0.0f;
  for (int l = 0; l < C_L; ++l) s += Vb[((size_t)(b * C_L + l)) * C_DM + h * C_DH + d];
  vm[bh * C_DH + d] = s * (1.0f / C_L);
}

__global__ void k_fillctx(const float* __restrict__ vm, h16* __restrict__ ctxh) {
  size_t i = (size_t)blockIdx.x * 256 + threadIdx.x;
  if (i >= (size_t)C_MTOT * C_DM) return;
  int c = (int)(i % C_DM);
  int bl = (int)(i / C_DM);
  int b = bl / C_L;
  int h = c >> 6, d = c & 63;
  ctxh[i] = (h16)vm[(b * C_H + h) * C_DH + d];
}

__global__ void k_scatter(const float* __restrict__ upd, const int* __restrict__ mtop,
                          h16* __restrict__ ctxh) {
  int i = blockIdx.x * 256 + threadIdx.x;
  if (i >= C_B * C_H * C_U * C_DH) return;
  int d = i & 63, t = i >> 6;
  int u = t % C_U, bh = t / C_U;
  int b = bh >> 3, h = bh & 7;
  int l = mtop[bh * C_U + u];
  ctxh[((size_t)(b * C_L + l)) * C_DM + h * C_DH + d] = (h16)upd[i];
}

// out = LN(a + b2) * g + bb  (row of 512); optional f16 copy
__global__ __launch_bounds__(256) void k_addln(
    const float* __restrict__ a, const float* __restrict__ b2,
    const float* __restrict__ g, const float* __restrict__ bb,
    float* __restrict__ out, h16* __restrict__ outh)
{
  __shared__ float red[256];
  int row = blockIdx.x, tid = threadIdx.x;
  const float* pa = a + (size_t)row * C_DM;
  const float* pb = b2 + (size_t)row * C_DM;
  float v0 = pa[tid] + pb[tid];
  float v1 = pa[tid + 256] + pb[tid + 256];
  red[tid] = v0 + v1; __syncthreads();
  for (int s = 128; s > 0; s >>= 1) { if (tid < s) red[tid] += red[tid + s]; __syncthreads(); }
  float mean = red[0] * (1.0f / C_DM); __syncthreads();
  red[tid] = (v0 - mean) * (v0 - mean) + (v1 - mean) * (v1 - mean); __syncthreads();
  for (int s = 128; s > 0; s >>= 1) { if (tid < s) red[tid] += red[tid + s]; __syncthreads(); }
  float inv = rsqrtf(red[0] * (1.0f / C_DM) + 1e-5f);
  float o0 = (v0 - mean) * inv * g[tid] + bb[tid];
  float o1 = (v1 - mean) * inv * g[tid + 256] + bb[tid + 256];
  size_t base = (size_t)row * C_DM;
  out[base + tid] = o0;
  out[base + tid + 256] = o1;
  if (outh) { outh[base + tid] = (h16)o0; outh[base + tid + 256] = (h16)o1; }
}

// im2col for circular conv1d: Aconv[bl, t*DM + i] = x2[b, (l-1+t) mod L, i]
__global__ void k_gatherconv(const float* __restrict__ x2, h16* __restrict__ d) {
  size_t i = (size_t)blockIdx.x * 256 + threadIdx.x;
  if (i >= (size_t)C_MTOT * 3 * C_DM) return;
  int col = (int)(i % (3 * C_DM));
  int bl = (int)(i / (3 * C_DM));
  int b = bl / C_L, l = bl % C_L;
  int t = col / C_DM, ii = col % C_DM;
  int ls = (l - 1 + t + C_L) & (C_L - 1);
  d[i] = (h16)x2[((size_t)(b * C_L + ls)) * C_DM + ii];
}

__global__ void k_bnelu(float* __restrict__ a, const float* __restrict__ g,
                        const float* __restrict__ b, const float* __restrict__ mean,
                        const float* __restrict__ var) {
  size_t i = (size_t)blockIdx.x * 256 + threadIdx.x;
  if (i >= (size_t)C_MTOT * C_DM) return;
  int o = (int)(i % C_DM);
  float v = (a[i] - mean[o]) * rsqrtf(var[o] + 1e-5f) * g[o] + b[o];
  a[i] = (v > 0.0f) ? v : (expf(v) - 1.0f);
}

// max-pool over batch dim, window 3 stride 2 pad 1 -> out[p, l, o], p in [0,8)
__global__ void k_pool(const float* __restrict__ act, float* __restrict__ out) {
  size_t i = (size_t)blockIdx.x * 256 + threadIdx.x;
  if (i >= (size_t)8 * C_L * C_DM) return;
  int o = (int)(i % C_DM);
  int rest = (int)(i / C_DM);
  int l = rest % C_L, p = rest / C_L;
  float m = -3.4e38f;
  for (int t = 0; t < 3; ++t) {
    int b = 2 * p - 1 + t;
    if (b >= 0 && b < C_B) m = fmaxf(m, act[((size_t)(b * C_L + l)) * C_DM + o]);
  }
  out[i] = m;
}

// ---------------------------------------------------------------------------
extern "C" void kernel_launch(void* const* d_in, const int* in_sizes, int n_in,
                              void* d_out, int out_size, void* d_ws, size_t ws_size,
                              hipStream_t stream) {
  (void)in_sizes; (void)n_in; (void)out_size;
  if (ws_size < WS_TOTAL) return;

  const float* x      = (const float*)d_in[0];
  const int*   idx    = (const int*)  d_in[1];
  const float* Wq     = (const float*)d_in[2];  const float* bq     = (const float*)d_in[3];
  const float* Wk     = (const float*)d_in[4];  const float* bk     = (const float*)d_in[5];
  const float* Wv     = (const float*)d_in[6];  const float* bv     = (const float*)d_in[7];
  const float* Wo     = (const float*)d_in[8];  const float* bo     = (const float*)d_in[9];
  const float* ln1g   = (const float*)d_in[10]; const float* ln1b   = (const float*)d_in[11];
  const float* conv1w = (const float*)d_in[12]; const float* conv1b = (const float*)d_in[13];
  const float* conv2w = (const float*)d_in[14]; const float* conv2b = (const float*)d_in[15];
  const float* ln2g   = (const float*)d_in[16]; const float* ln2b   = (const float*)d_in[17];
  const float* downw  = (const float*)d_in[18]; const float* downb  = (const float*)d_in[19];
  const float* bng    = (const float*)d_in[20]; const float* bnb    = (const float*)d_in[21];
  const float* bnmean = (const float*)d_in[22]; const float* bnvar  = (const float*)d_in[23];
  float* out = (float*)d_out;

  char* w = (char*)d_ws;
  h16*   xh      = (h16*)  (w + OFF_XH);
  h16*   wh      = (h16*)  (w + OFF_WH);
  float* Qf      = (float*)(w + OFF_Q);
  float* Kf      = (float*)(w + OFF_K);
  float* Vf      = (float*)(w + OFF_V);
  float* Mb      = (float*)(w + OFF_M);
  int*   mtop    = (int*)  (w + OFF_MTOP);
  float* upd     = (float*)(w + OFF_UPD);
  float* vmean   = (float*)(w + OFF_VMEAN);
  h16*   ctxh    = (h16*)  (w + OFF_CTXH);
  float* attnout = (float*)(w + OFF_ATTNOUT);
  float* x1      = (float*)(w + OFF_X1);
  h16*   x1h     = (h16*)  (w + OFF_X1H);
  // reused regions
  h16*   yh      = (h16*)  (w + OFF_Q);
  float* ffn2out = (float*)(w + OFF_V);
  float* x2      = (float*)(w + OFF_ATTNOUT);
  h16*   Aconv   = (h16*)  (w + OFF_Q);
  float* convout = (float*)(w + OFF_X1);

  const size_t n_x = (size_t)C_MTOT * C_DM;
  const size_t n_w = (size_t)C_DM * C_DM;
  const dim3 blk(256);
  auto gridN = [](size_t n) { return dim3((unsigned)((n + 255) / 256)); };

  dim3 g512(C_MTOT / BLK_M, C_DM / BLK_N);    // (128, 4)
  dim3 g2048(C_MTOT / BLK_M, C_DFF / BLK_N);  // (128, 16)

  // ---- phase 1: Q/K/V projections (WMMA); weights pre-transposed to (N,K) ----
  k_cvt<<<gridN(n_x), blk, 0, stream>>>(x, xh, n_x);
  k_cvt_tr<<<gridN(n_w), blk, 0, stream>>>(Wq, wh, C_DM, C_DM);
  k_gemm<<<g512, blk, 0, stream>>>(xh, wh, bq, Qf, (h16*)nullptr, C_DM, C_DM, 0);
  k_cvt_tr<<<gridN(n_w), blk, 0, stream>>>(Wk, wh, C_DM, C_DM);
  k_gemm<<<g512, blk, 0, stream>>>(xh, wh, bk, Kf, (h16*)nullptr, C_DM, C_DM, 0);
  k_cvt_tr<<<gridN(n_w), blk, 0, stream>>>(Wv, wh, C_DM, C_DM);
  k_gemm<<<g512, blk, 0, stream>>>(xh, wh, bv, Vf, (h16*)nullptr, C_DM, C_DM, 0);

  // ---- phase 2: ProbSparse attention ----
  k_probe<<<dim3(C_B * C_H * C_L / 256), blk, 0, stream>>>(Qf, Kf, idx, Mb);
  k_topk<<<dim3(C_B * C_H), blk, 0, stream>>>(Mb, mtop);
  k_attn<<<dim3(C_B * C_H * C_U), blk, 0, stream>>>(Qf, Kf, Vf, mtop, upd);
  k_vmean<<<dim3(C_B * C_H), dim3(C_DH), 0, stream>>>(Vf, vmean);
  k_fillctx<<<gridN(n_x), blk, 0, stream>>>(vmean, ctxh);
  k_scatter<<<gridN((size_t)C_B * C_H * C_U * C_DH), blk, 0, stream>>>(upd, mtop, ctxh);

  // ---- phase 3: output projection + LN1 ----
  k_cvt_tr<<<gridN(n_w), blk, 0, stream>>>(Wo, wh, C_DM, C_DM);
  k_gemm<<<g512, blk, 0, stream>>>(ctxh, wh, bo, attnout, (h16*)nullptr, C_DM, C_DM, 0);
  k_addln<<<dim3(C_MTOT), blk, 0, stream>>>(x, attnout, ln1g, ln1b, x1, x1h);

  // ---- phase 4: FFN (conv1_w / conv2_w are already (N,K) row-major) ----
  k_cvt<<<gridN((size_t)C_DFF * C_DM), blk, 0, stream>>>(conv1w, wh, (size_t)C_DFF * C_DM);
  k_gemm<<<g2048, blk, 0, stream>>>(x1h, wh, conv1b, (float*)nullptr, yh, C_DFF, C_DM, 1);
  k_cvt<<<gridN((size_t)C_DM * C_DFF), blk, 0, stream>>>(conv2w, wh, (size_t)C_DM * C_DFF);
  k_gemm<<<g512, blk, 0, stream>>>(yh, wh, conv2b, ffn2out, (h16*)nullptr, C_DM, C_DFF, 0);
  k_addln<<<dim3(C_MTOT), blk, 0, stream>>>(x1, ffn2out, ln2g, ln2b, x2, (h16*)nullptr);

  // ---- phase 5: circular conv1d as im2col GEMM (K=1536) ----
  k_gatherconv<<<gridN((size_t)C_MTOT * 3 * C_DM), blk, 0, stream>>>(x2, Aconv);
  k_cvt_downw<<<gridN((size_t)C_DM * C_DM * 3), blk, 0, stream>>>(downw, wh);
  k_gemm<<<g512, blk, 0, stream>>>(Aconv, wh, downb, convout, (h16*)nullptr, C_DM, 3 * C_DM, 0);

  // ---- phase 6: BN + ELU, then batch max-pool ----
  k_bnelu<<<gridN(n_x), blk, 0, stream>>>(convout, bng, bnb, bnmean, bnvar);
  k_pool<<<gridN((size_t)8 * C_L * C_DM), blk, 0, stream>>>(convout, out);
}
